// FusionBlock_62895501082940
// MI455X (gfx1250) — compile-verified, matrix operand order checked by reading, not run on hardware
//
#include <hip/hip_runtime.h>
#include <hip/hip_bf16.h>

typedef __attribute__((ext_vector_type(16))) _Float16 v16h;
typedef __attribute__((ext_vector_type(8)))  _Float16 v8h;
typedef __attribute__((ext_vector_type(8)))  float    v8f;

#define DEV_INLINE __device__ __forceinline__

namespace {
enum : int { B = 4, N = 256, DX = 256, DE = 64, DY = 128, NH = 8, DF = 32,
             GK = 128, XK = 64 };
enum : int { NN = N * N };
constexpr float LN_EPS = 1e-6f;
}

// ---------------------------------------------------------------------------
// CDNA5 16-bit fragment: lane (half,lr) holds K = k0 + half*8 + {0..7} in
// elements 0..7 and K = k0 + 16 + half*8 + {0..7} in elements 8..15, i.e. two
// contiguous 16B-aligned 8-half groups -> two global_load_b128.
// ---------------------------------------------------------------------------
DEV_INLINE v16h load_frag(const _Float16* __restrict__ p, int half)
{
    const v8h lo = *(const v8h*)(p + half * 8);
    const v8h hi = *(const v8h*)(p + 16 + half * 8);
    return __builtin_shufflevector(lo, hi, 0, 1, 2, 3, 4, 5, 6, 7,
                                   8, 9, 10, 11, 12, 13, 14, 15);
}

// ---------------------------------------------------------------------------
// f16 x f16^T batched GEMM: C[z] = A[z] (MxK, f16 row-major) * Bt[z]^T
// (Bt is KxN pre-transposed to NxK f16 row-major). One wave computes a 32x32
// C tile: 8 b128 loads + 4 independent v_wmma_f32_16x16x32_f16 per K-step,
// no conversions in the loop. Requires M%32==0, N%32==0, K%32==0, lda/ldbt%8==0.
// Used for the dominant GEMMs (Gr polynomial, newE@W_Eout).
// ---------------------------------------------------------------------------
__global__ void wmma_gemm_f16t(const _Float16* __restrict__ A,
                               const _Float16* __restrict__ Bt,
                               float* __restrict__ C,
                               int K, int lda, int ldbt, int ldc,
                               long long sA, long long sB, long long sC)
{
    A  += blockIdx.z * sA;
    Bt += blockIdx.z * sB;
    C  += blockIdx.z * sC;

    const int lane = threadIdx.x & 31;
    const int half = lane >> 4;
    const int lr   = lane & 15;
    const int m0 = blockIdx.y << 5;
    const int n0 = blockIdx.x << 5;

    const _Float16* a0p = A  + (long long)(m0 + lr)      * lda;
    const _Float16* a1p = A  + (long long)(m0 + 16 + lr) * lda;
    const _Float16* b0p = Bt + (long long)(n0 + lr)      * ldbt;
    const _Float16* b1p = Bt + (long long)(n0 + 16 + lr) * ldbt;

    v8f acc00 = {}, acc01 = {}, acc10 = {}, acc11 = {};

    for (int k0 = 0; k0 < K; k0 += 32) {
        const v16h a0 = load_frag(a0p + k0, half);
        const v16h a1 = load_frag(a1p + k0, half);
        const v16h b0 = load_frag(b0p + k0, half);
        const v16h b1 = load_frag(b1p + k0, half);
        acc00 = __builtin_amdgcn_wmma_f32_16x16x32_f16(false, a0, false, b0,
                                                       (short)0, acc00, false, false);
        acc01 = __builtin_amdgcn_wmma_f32_16x16x32_f16(false, a0, false, b1,
                                                       (short)0, acc01, false, false);
        acc10 = __builtin_amdgcn_wmma_f32_16x16x32_f16(false, a1, false, b0,
                                                       (short)0, acc10, false, false);
        acc11 = __builtin_amdgcn_wmma_f32_16x16x32_f16(false, a1, false, b1,
                                                       (short)0, acc11, false, false);
    }

#pragma unroll
    for (int r = 0; r < 8; ++r) {
        const int mr0 = m0 + half * 8 + r;     // C layout: VGPR r -> row r (+8 hi)
        float* cp0 = C + (long long)mr0 * ldc + n0 + lr;
        cp0[0]  = acc00[r];
        cp0[16] = acc01[r];
        float* cp1 = C + (long long)(mr0 + 16) * ldc + n0 + lr;
        cp1[0]  = acc10[r];
        cp1[16] = acc11[r];
    }
}

// ---------------------------------------------------------------------------
// Generic batched f32 WMMA GEMM (32x32 tile per wave, converts in regs).
// TRANSB/MGUARD compile-time; MGUARD=1 only for the tiny M=4 GEMMs.
// ---------------------------------------------------------------------------
template <int TRANSB, int MGUARD>
__global__ void wmma_gemm_kernel(const float* __restrict__ A,
                                 const float* __restrict__ Bm,
                                 float* __restrict__ C,
                                 int M, int K,
                                 int lda, int ldb, int ldc,
                                 int zdiv,
                                 long long sAb, long long sAh,
                                 long long sBb, long long sBh,
                                 long long sCb, long long sCh)
{
    const int z  = blockIdx.z;
    const int zb = z / zdiv, zh = z % zdiv;
    A  += zb * sAb + zh * sAh;
    Bm += zb * sBb + zh * sBh;
    C  += zb * sCb + zh * sCh;

    const int lane = threadIdx.x & 31;
    const int half = lane >> 4;
    const int lr   = lane & 15;
    const int m0 = blockIdx.y << 5;
    const int n0 = blockIdx.x << 5;

    const int mA0 = m0 + lr,      mA1 = m0 + 16 + lr;
    const int nB0 = n0 + lr,      nB1 = n0 + 16 + lr;

    v8f acc00 = {}, acc01 = {}, acc10 = {}, acc11 = {};

    for (int k0 = 0; k0 < K; k0 += 32) {
        v16h a0, a1, b0, b1;
#pragma unroll
        for (int v = 0; v < 8; ++v) {
            const int kk = k0 + ((v < 4) ? (half * 8 + 2 * v)
                                         : (16 + half * 8 + 2 * (v - 4)));
            float x0 = 0.f, x1 = 0.f, x2 = 0.f, x3 = 0.f;
            if (!MGUARD || mA0 < M) {
                const float* p = A + (long long)mA0 * lda + kk;
                x0 = p[0]; x1 = p[1];
            }
            if (!MGUARD || mA1 < M) {
                const float* p = A + (long long)mA1 * lda + kk;
                x2 = p[0]; x3 = p[1];
            }
            a0[2 * v] = (_Float16)x0; a0[2 * v + 1] = (_Float16)x1;
            a1[2 * v] = (_Float16)x2; a1[2 * v + 1] = (_Float16)x3;

            float y0, y1, y2, y3;
            if (TRANSB) {
                const float* p0 = Bm + (long long)nB0 * ldb + kk;
                const float* p1 = Bm + (long long)nB1 * ldb + kk;
                y0 = p0[0]; y1 = p0[1];
                y2 = p1[0]; y3 = p1[1];
            } else {
                const float* p0 = Bm + (long long)kk * ldb;
                const float* p1 = Bm + (long long)(kk + 1) * ldb;
                y0 = p0[nB0]; y1 = p1[nB0];
                y2 = p0[nB1]; y3 = p1[nB1];
            }
            b0[2 * v] = (_Float16)y0; b0[2 * v + 1] = (_Float16)y1;
            b1[2 * v] = (_Float16)y2; b1[2 * v + 1] = (_Float16)y3;
        }
        acc00 = __builtin_amdgcn_wmma_f32_16x16x32_f16(false, a0, false, b0,
                                                       (short)0, acc00, false, false);
        acc01 = __builtin_amdgcn_wmma_f32_16x16x32_f16(false, a0, false, b1,
                                                       (short)0, acc01, false, false);
        acc10 = __builtin_amdgcn_wmma_f32_16x16x32_f16(false, a1, false, b0,
                                                       (short)0, acc10, false, false);
        acc11 = __builtin_amdgcn_wmma_f32_16x16x32_f16(false, a1, false, b1,
                                                       (short)0, acc11, false, false);
    }

#pragma unroll
    for (int r = 0; r < 8; ++r) {
        const int mr0 = m0 + half * 8 + r;
        const int mr1 = mr0 + 16;
        if (!MGUARD || mr0 < M) {
            float* cp = C + (long long)mr0 * ldc + n0 + lr;
            cp[0]  = acc00[r];
            cp[16] = acc01[r];
        }
        if (!MGUARD || mr1 < M) {
            float* cp = C + (long long)mr1 * ldc + n0 + lr;
            cp[0]  = acc10[r];
            cp[16] = acc11[r];
        }
    }
}

static inline void gemm(hipStream_t st, const float* A, const float* Bm, float* C,
                        int M, int Nn, int K, int lda, int ldb, int ldc,
                        int transB = 0, int batches = 1, int zdiv = 1,
                        long long sAb = 0, long long sAh = 0,
                        long long sBb = 0, long long sBh = 0,
                        long long sCb = 0, long long sCh = 0)
{
    dim3 grid(Nn / 32, (M + 31) / 32, batches);
    const bool guard = (M & 31) != 0;
    if (transB) {
        if (guard)
            wmma_gemm_kernel<1, 1><<<grid, dim3(32), 0, st>>>(A, Bm, C, M, K,
                lda, ldb, ldc, zdiv, sAb, sAh, sBb, sBh, sCb, sCh);
        else
            wmma_gemm_kernel<1, 0><<<grid, dim3(32), 0, st>>>(A, Bm, C, M, K,
                lda, ldb, ldc, zdiv, sAb, sAh, sBb, sBh, sCb, sCh);
    } else {
        if (guard)
            wmma_gemm_kernel<0, 1><<<grid, dim3(32), 0, st>>>(A, Bm, C, M, K,
                lda, ldb, ldc, zdiv, sAb, sAh, sBb, sBh, sCb, sCh);
        else
            wmma_gemm_kernel<0, 0><<<grid, dim3(32), 0, st>>>(A, Bm, C, M, K,
                lda, ldb, ldc, zdiv, sAb, sAh, sBb, sBh, sCb, sCh);
    }
}

// ---------------------------------------------------------------------------
// Block-wide reductions (blockDim.x must be a power of two)
// ---------------------------------------------------------------------------
DEV_INLINE float block_reduce_sum(float v, float* sh) {
    const int t = threadIdx.x;
    sh[t] = v; __syncthreads();
    for (int s = blockDim.x >> 1; s > 0; s >>= 1) {
        if (t < s) sh[t] += sh[t + s];
        __syncthreads();
    }
    const float r = sh[0];
    __syncthreads();
    return r;
}
DEV_INLINE float block_reduce_max(float v, float* sh) {
    const int t = threadIdx.x;
    sh[t] = v; __syncthreads();
    for (int s = blockDim.x >> 1; s > 0; s >>= 1) {
        if (t < s) sh[t] = fmaxf(sh[t], sh[t + s]);
        __syncthreads();
    }
    const float r = sh[0];
    __syncthreads();
    return r;
}

// ---------------------------------------------------------------------------
// f32 -> f16 conversion (flat) and per-slab transposed conversion.
// ---------------------------------------------------------------------------
__global__ void k_cvt_f16(const float* __restrict__ in, _Float16* __restrict__ out,
                          long long n)
{
    long long idx = (long long)blockIdx.x * blockDim.x + threadIdx.x;
    if (idx < n) out[idx] = (_Float16)in[idx];
}

// out[z][c][r] = (f16) in[z][r][c]   for `total` = slabs*rows*cols elements
__global__ void k_cvt_f16_T(const float* __restrict__ in, _Float16* __restrict__ out,
                            int rows, int cols, long long total)
{
    long long idx = (long long)blockIdx.x * blockDim.x + threadIdx.x;
    if (idx >= total) return;
    const int slabsz = rows * cols;
    const long long z = idx / slabsz;
    const int rem = (int)(idx - z * slabsz);
    const int r = rem / cols, c = rem % cols;
    out[z * slabsz + (long long)c * rows + r] = (_Float16)in[idx];
}

// ---------------------------------------------------------------------------
// Ebar[b,i,d] = m_i * (1/N) * sum_j m_j * E[b,i,j,d]
// (mean_j((E@W)*mask) == (mean_j E*mask)@W; the GEMM is done afterwards)
// ---------------------------------------------------------------------------
__global__ void k_ebar(const float* __restrict__ E, const int* __restrict__ nm,
                       float* __restrict__ Ebar)
{
    int idx = blockIdx.x * blockDim.x + threadIdx.x;
    if (idx >= B * N * DE) return;
    const int d = idx % DE;
    const int i = (idx / DE) % N;
    const int b = idx / (DE * N);
    const float mi = nm[b * N + i] ? 1.f : 0.f;
    float s = 0.f;
    const float* base = E + ((long long)(b * N + i) * N) * DE + d;
    for (int j = 0; j < N; ++j)
        s += base[(long long)j * DE] * (nm[b * N + j] ? 1.f : 0.f);
    Ebar[idx] = s * mi * (1.f / N);
}

__global__ void k_relu(float* __restrict__ p, long long n) {
    long long idx = (long long)blockIdx.x * blockDim.x + threadIdx.x;
    if (idx < n) p[idx] = fmaxf(p[idx], 0.f);
}

// xplus = (X + relu(t1)) * x_mask
__global__ void k_xplus(const float* __restrict__ X, const float* __restrict__ t1,
                        const int* __restrict__ nm, float* __restrict__ out)
{
    int idx = blockIdx.x * blockDim.x + threadIdx.x;
    if (idx >= B * N * DX) return;
    const int row = idx / DX;                  // b*N + i
    const float m = nm[row] ? 1.f : 0.f;
    out[idx] = (X[idx] + fmaxf(t1[idx], 0.f)) * m;
}

// out[b,i,:] = X[b,i,:] + yb[b,:]
__global__ void k_add_bcast(const float* __restrict__ X,
                            const float* __restrict__ yb,
                            float* __restrict__ out)
{
    int idx = blockIdx.x * blockDim.x + threadIdx.x;
    if (idx >= B * N * DX) return;
    const int b = idx / (N * DX);
    const int c = idx % DX;
    out[idx] = X[idx] + yb[b * DX + c];
}

// Row softmax over keys with node mask and 1/sqrt(DF) scale (in place).
__global__ void k_softmax_attn(float* __restrict__ scores,
                               const int* __restrict__ nm)
{
    __shared__ float red[N];
    const int row = blockIdx.x;                // (b*NH + h)*N + q
    const int b = row / (NH * N);
    float* s = scores + (long long)row * N;
    const int t = threadIdx.x;
    const float l = nm[b * N + t] ? s[t] * 0.17677669529663687f : -1e9f;
    const float gmax = block_reduce_max(l, red);
    const float e = __expf(l - gmax);
    const float gsum = block_reduce_sum(e, red);
    s[t] = e / gsum;
}

// new_X = LayerNorm(t2 * x_mask + X) over DX
__global__ void k_ln_rowX(const float* __restrict__ t2,
                          const float* __restrict__ X,
                          const int* __restrict__ nm,
                          const float* __restrict__ g,
                          const float* __restrict__ bsh,
                          float* __restrict__ out)
{
    __shared__ float red[DX];
    const int row = blockIdx.x;                // b*N + i
    const int t = threadIdx.x;
    const float m = nm[row] ? 1.f : 0.f;
    const long long off = (long long)row * DX + t;
    const float v = t2[off] * m + X[off];
    const float mean = block_reduce_sum(v, red) * (1.f / DX);
    const float dv = v - mean;
    const float var = block_reduce_sum(dv * dv, red) * (1.f / DX);
    out[off] = dv * rsqrtf(var + LN_EPS) * g[t] + bsh[t];
}

// G = (alpha * sim/sqrt(DX) + E) * emask
__global__ void k_build_G(const float* __restrict__ E,
                          const float* __restrict__ sim,
                          const int* __restrict__ nm,
                          const float* __restrict__ alpha,
                          float* __restrict__ G)
{
    int idx = blockIdx.x * blockDim.x + threadIdx.x;
    if (idx >= B * NN * DE) return;
    const int j = (idx / DE) % N;
    const int i = (idx / (DE * N)) % N;
    const int b = idx / (DE * NN);
    const float em = (nm[b * N + i] && nm[b * N + j]) ? 1.f : 0.f;
    const float sv = sim[((long long)b * N + i) * N + j] * 0.0625f;
    G[idx] = (alpha[0] * sv + E[idx]) * em;
}

// coeff[b,o] = y[b] . Wc[:,o] + bc[o]   (12 values)
__global__ void k_coeff(const float* __restrict__ y, const float* __restrict__ Wc,
                        const float* __restrict__ bc, float* __restrict__ coeff)
{
    const int t = threadIdx.x;
    if (t >= B * 3) return;
    const int b = t / 3, o = t % 3;
    float s = bc[o];
    for (int k = 0; k < DY; ++k) s += y[b * DY + k] * Wc[k * 3 + o];
    coeff[t] = s;
}

// P = c0[b]*P + c1[b]*G      (newE after first Horner step, in place)
__global__ void k_polyA(float* __restrict__ P, const float* __restrict__ G,
                        const float* __restrict__ coeff)
{
    int idx = blockIdx.x * blockDim.x + threadIdx.x;
    if (idx >= B * NN * DE) return;
    const int b = idx / (NN * DE);
    P[idx] = coeff[b * 3 + 0] * P[idx] + coeff[b * 3 + 1] * G[idx];
}

// P += c2[b]*G               (final newE, in place)
__global__ void k_polyB(float* __restrict__ P, const float* __restrict__ G,
                        const float* __restrict__ coeff)
{
    int idx = blockIdx.x * blockDim.x + threadIdx.x;
    if (idx >= B * NN * DE) return;
    const int b = idx / (NN * DE);
    P[idx] += coeff[b * 3 + 2] * G[idx];
}

// new_E = LayerNorm(relu(t4)*emask + E) over DE; in place on t4 (d_out region)
__global__ void k_newE_ln(float* __restrict__ t4io, const float* __restrict__ E,
                          const int* __restrict__ nm,
                          const float* __restrict__ ge,
                          const float* __restrict__ be)
{
    __shared__ float red[DE];
    const int r = blockIdx.x;                  // b*NN + p
    const int t = threadIdx.x;                 // 0..63
    const int p = r % NN;
    const int b = r / NN;
    const int i = p / N, j = p % N;
    const float em = (nm[b * N + i] && nm[b * N + j]) ? 1.f : 0.f;
    const long long off = (long long)r * DE + t;
    float v = fmaxf(t4io[off], 0.f) * em + E[off];
    const float mean = block_reduce_sum(v, red) * (1.f / DE);
    const float dv = v - mean;
    const float var = block_reduce_sum(dv * dv, red) * (1.f / DE);
    t4io[off] = dv * rsqrtf(var + LN_EPS) * ge[t] + be[t];
}

// G_enc[b,k,:] = softmax_p(lg[b,p,k]) . Em[b,p,:]  with online recompute,
// never materializing the (B,NN,GK) logit tensor. One block per (b,k).
__global__ void k_g_enc(const float* __restrict__ E, const int* __restrict__ nm,
                        const float* __restrict__ Gker, float* __restrict__ G_enc)
{
    __shared__ float red[256];
    __shared__ float wsh[256];
    const int b = blockIdx.x / GK;
    const int k = blockIdx.x % GK;
    const int t = threadIdx.x;
    const long long ebase = (long long)b * NN * DE;

    // pass 1: global max of logits
    float lmax = -1e30f;
    for (int p = t; p < NN; p += 256) {
        const int i = p >> 8, j = p & 255;
        float l = -1e9f;
        if (nm[b * N + i] && nm[b * N + j]) {
            const float* er = E + ebase + (long long)p * DE;
            float s = 0.f;
            for (int d = 0; d < DE; ++d) s += er[d] * Gker[d * GK + k];
            l = s * 0.125f;
        }
        lmax = fmaxf(lmax, l);
    }
    const float gmax = block_reduce_max(lmax, red);

    // pass 2: sum of exp
    float lsum = 0.f;
    for (int p = t; p < NN; p += 256) {
        const int i = p >> 8, j = p & 255;
        float l = -1e9f;
        if (nm[b * N + i] && nm[b * N + j]) {
            const float* er = E + ebase + (long long)p * DE;
            float s = 0.f;
            for (int d = 0; d < DE; ++d) s += er[d] * Gker[d * GK + k];
            l = s * 0.125f;
        }
        lsum += __expf(l - gmax);
    }
    const float gsum = block_reduce_sum(lsum, red);

    // pass 3: weighted accumulation of Em rows (mask folded into weight)
    const int dg = t & 63;
    const int grp = t >> 6;                    // 4 groups of 64 d-lanes
    float acc = 0.f;
    for (int base = 0; base < NN; base += 256) {
        const int p = base + t;
        const int i = p >> 8, j = p & 255;
        float w = 0.f;
        if (nm[b * N + i] && nm[b * N + j]) {
            const float* er = E + ebase + (long long)p * DE;
            float s = 0.f;
            for (int d = 0; d < DE; ++d) s += er[d] * Gker[d * GK + k];
            w = __expf(s * 0.125f - gmax);
        }
        wsh[t] = w;
        __syncthreads();
        const int q0 = grp * 64;
        for (int q = 0; q < 64; ++q)
            acc += wsh[q0 + q] * E[ebase + (long long)(base + q0 + q) * DE + dg];
        __syncthreads();
    }
    red[t] = acc;
    __syncthreads();
    if (grp == 0) {
        const float s = red[dg] + red[64 + dg] + red[128 + dg] + red[192 + dg];
        G_enc[((long long)b * GK + k) * DE + dg] = s / gsum;
    }
}

// X_enc[b,k,:] = softmax_n(lx[b,n,k]) . Xm[b,n,:]. One block per (b,k).
__global__ void k_x_enc(const float* __restrict__ X, const int* __restrict__ nm,
                        const float* __restrict__ Xker, float* __restrict__ X_enc)
{
    __shared__ float red[N];
    __shared__ float lsh[N];
    const int b = blockIdx.x / XK;
    const int k = blockIdx.x % XK;
    const int t = threadIdx.x;                 // n for logits, d for accum

    const float mn = nm[b * N + t] ? 1.f : 0.f;
    float l = -1e9f;
    if (mn > 0.f) {
        const float* xr = X + ((long long)b * N + t) * DX;
        float s = 0.f;
        for (int d = 0; d < DX; ++d) s += xr[d] * Xker[d * XK + k];
        l = s * 0.0625f;
    }
    const float gmax = block_reduce_max(l, red);
    const float w = __expf(l - gmax);
    const float gsum = block_reduce_sum(w, red);
    lsh[t] = w * mn;
    __syncthreads();

    float acc = 0.f;
    for (int n = 0; n < N; ++n)
        acc += lsh[n] * X[((long long)b * N + n) * DX + t];
    X_enc[((long long)b * XK + k) * DX + t] = acc / gsum;
}

// h = relu(g1) + relu(x1)
__global__ void k_combine(const float* __restrict__ g1,
                          const float* __restrict__ x1,
                          float* __restrict__ h)
{
    int idx = blockIdx.x * blockDim.x + threadIdx.x;
    if (idx >= B * DY) return;
    h[idx] = fmaxf(g1[idx], 0.f) + fmaxf(x1[idx], 0.f);
}

// new_y = LayerNorm(relu(t5) + y) over DY
__global__ void k_newy_ln(const float* __restrict__ t5, const float* __restrict__ y,
                          const float* __restrict__ gy, const float* __restrict__ by,
                          float* __restrict__ out)
{
    __shared__ float red[DY];
    const int b = blockIdx.x;
    const int t = threadIdx.x;
    const float v = fmaxf(t5[b * DY + t], 0.f) + y[b * DY + t];
    const float mean = block_reduce_sum(v, red) * (1.f / DY);
    const float dv = v - mean;
    const float var = block_reduce_sum(dv * dv, red) * (1.f / DY);
    out[b * DY + t] = dv * rsqrtf(var + LN_EPS) * gy[t] + by[t];
}

// ---------------------------------------------------------------------------
extern "C" void kernel_launch(void* const* d_in, const int* in_sizes, int n_in,
                              void* d_out, int out_size, void* d_ws, size_t ws_size,
                              hipStream_t stream)
{
    (void)in_sizes; (void)n_in; (void)out_size; (void)ws_size;

    const float* X       = (const float*)d_in[0];
    const float* E       = (const float*)d_in[1];
    const float* y       = (const float*)d_in[2];
    const int*   nm      = (const int*)  d_in[3];
    const float* W_emax  = (const float*)d_in[4];
    const float* W_x1    = (const float*)d_in[5];
    const float* Wq      = (const float*)d_in[6];
    const float* Wk      = (const float*)d_in[7];
    const float* Wv      = (const float*)d_in[8];
    const float* Wo      = (const float*)d_in[9];
    const float* W_y2    = (const float*)d_in[10];
    const float* W_x2    = (const float*)d_in[11];
    const float* Wc      = (const float*)d_in[12];
    const float* bc      = (const float*)d_in[13];
    const float* alpha   = (const float*)d_in[14];
    const float* W_Eout  = (const float*)d_in[15];
    const float* G_ker   = (const float*)d_in[16];
    const float* X_ker   = (const float*)d_in[17];
    const float* W_Gtr   = (const float*)d_in[18];
    const float* W_Xtr   = (const float*)d_in[19];
    const float* W_yout  = (const float*)d_in[20];
    const float* gx      = (const float*)d_in[21];
    const float* bx      = (const float*)d_in[22];
    const float* ge      = (const float*)d_in[23];
    const float* be      = (const float*)d_in[24];
    const float* gy      = (const float*)d_in[25];
    const float* by      = (const float*)d_in[26];

    float* wsf = (float*)d_ws;
    size_t o = 0;
    auto nxt = [&](size_t n) { float* p = wsf + o; o += n; return p; };
    float* Ebar   = nxt((size_t)B * N * DE);
    float* e_enc  = nxt((size_t)B * N * DX);
    float* t1     = nxt((size_t)B * N * DX);
    float* xplus  = nxt((size_t)B * N * DX);
    float* qb     = nxt((size_t)B * N * DX);
    float* kb     = nxt((size_t)B * N * DX);
    float* vb     = nxt((size_t)B * N * DX);
    float* scores = nxt((size_t)B * NH * NN);
    float* attn   = nxt((size_t)B * N * DX);
    float* t2     = nxt((size_t)B * N * DX);
    float* yb     = nxt((size_t)B * DX);
    float* xb     = nxt((size_t)B * N * DX);
    float* xpy    = nxt((size_t)B * N * DX);
    float* simb   = nxt((size_t)B * NN);
    float* G      = nxt((size_t)B * NN * DE);
    float* P1     = nxt((size_t)B * NN * DE);
    float* P2     = nxt((size_t)B * NN * DE);
    float* coeff  = nxt(16);
    float* Genc   = nxt((size_t)B * GK * DE);
    float* Xenc   = nxt((size_t)B * XK * DX);
    float* g1     = nxt((size_t)B * DY);
    float* x1     = nxt((size_t)B * DY);
    float* hbuf   = nxt((size_t)B * DY);
    float* t5     = nxt((size_t)B * DY);
    // f16 staging buffers (sizes in float units = halves/2; all 16B aligned)
    _Float16* Gh  = (_Float16*)nxt((size_t)B * NN * DE / 2);   // f16(G)
    _Float16* BTh = (_Float16*)nxt((size_t)B * NN * DE / 2);   // f16(G^T) then f16(newE1^T)
    _Float16* P2h = (_Float16*)nxt((size_t)B * NN * DE / 2);   // f16(newE)
    _Float16* WEt = (_Float16*)nxt((size_t)(DE * DE) / 2 + 8); // f16(W_Eout^T)
    (void)o;

    float* out_X = (float*)d_out;                       // (B,N,DX)
    float* out_E = out_X + (size_t)B * N * DX;          // (B,N,N,DE)
    float* out_y = out_E + (size_t)B * NN * DE;         // (B,DY)

    auto ew = [](long long n) { return dim3((unsigned)((n + 255) / 256)); };
    const long long NXD = (long long)B * N * DX;        // 262144
    const long long NED = (long long)B * NN * DE;       // 16777216

    // ---- edge-feature encoding + xplus --------------------------------
    k_ebar<<<ew(B * N * DE), 256, 0, stream>>>(E, nm, Ebar);
    gemm(stream, Ebar, W_emax, e_enc, B * N, DX, DE, DE, DX, DX);
    k_relu<<<ew(NXD), 256, 0, stream>>>(e_enc, NXD);
    gemm(stream, e_enc, W_x1, t1, B * N, DX, DX, DX, DX, DX);
    k_xplus<<<ew(NXD), 256, 0, stream>>>(X, t1, nm, xplus);

    // ---- attention -----------------------------------------------------
    gemm(stream, xplus, Wq, qb, B * N, DX, DX, DX, DX, DX);
    gemm(stream, xplus, Wk, kb, B * N, DX, DX, DX, DX, DX);
    gemm(stream, xplus, Wv, vb, B * N, DX, DX, DX, DX, DX);
    // scores[b,h] = q[b,:,h] * k[b,:,h]^T   (B*NH batches, K=DF)
    gemm(stream, qb, kb, scores, N, N, DF, DX, DX, N,
         /*transB=*/1, B * NH, NH,
         (long long)N * DX, DF, (long long)N * DX, DF,
         (long long)NH * NN, NN);
    k_softmax_attn<<<B * NH * N, N, 0, stream>>>(scores, nm);
    // attn[b,:,h] = P[b,h] * v[b,:,h]
    gemm(stream, scores, vb, attn, N, DF, N, N, DX, DX,
         /*transB=*/0, B * NH, NH,
         (long long)NH * NN, NN, (long long)N * DX, DF,
         (long long)N * DX, DF);
    gemm(stream, attn, Wo, t2, B * N, DX, DX, DX, DX, DX);
    k_ln_rowX<<<B * N, DX, 0, stream>>>(t2, X, nm, gx, bx, out_X);

    // ---- xplusy / sim / G ---------------------------------------------
    gemm(stream, y, W_y2, yb, B, DX, DY, DY, DX, DX);
    k_relu<<<ew(B * DX), 256, 0, stream>>>(yb, B * DX);
    k_add_bcast<<<ew(NXD), 256, 0, stream>>>(X, yb, xb);
    gemm(stream, xb, W_x2, xpy, B * N, DX, DX, DX, DX, DX);
    k_relu<<<ew(NXD), 256, 0, stream>>>(xpy, NXD);
    gemm(stream, xpy, xpy, simb, N, N, DX, DX, DX, N,
         /*transB=*/1, B, 1,
         (long long)N * DX, 0, (long long)N * DX, 0, (long long)NN, 0);
    k_build_G<<<ew(NED), 256, 0, stream>>>(E, simb, nm, alpha, G);
    k_coeff<<<1, 32, 0, stream>>>(y, Wc, bc, coeff);

    // ---- matrix polynomial: newE = c0*Gr^3 + c1*Gr^2 + c2*Gr ----------
    // Gr[b,d] is a row-major 256x256 slab at offset (b*DE+d)*NN (reshape).
    // Stage f16 operands once: Gh = f16(G), BTh = f16(G^T); the hot loops
    // then run pure b128 loads + wmma with no conversions.
    k_cvt_f16  <<<ew(NED), 256, 0, stream>>>(G, Gh, NED);
    k_cvt_f16_T<<<ew(NED), 256, 0, stream>>>(G, BTh, N, N, NED);
    {
        dim3 g(N / 32, N / 32, B * DE);
        wmma_gemm_f16t<<<g, 32, 0, stream>>>(Gh, BTh, P1, N, N, N, N,
                                             (long long)NN, (long long)NN,
                                             (long long)NN);
    }
    k_polyA<<<ew(NED), 256, 0, stream>>>(P1, G, coeff);       // P1 = newE1 (f32)
    k_cvt_f16_T<<<ew(NED), 256, 0, stream>>>(P1, BTh, N, N, NED);
    {
        dim3 g(N / 32, N / 32, B * DE);
        wmma_gemm_f16t<<<g, 32, 0, stream>>>(Gh, BTh, P2, N, N, N, N,
                                             (long long)NN, (long long)NN,
                                             (long long)NN);
    }
    k_polyB<<<ew(NED), 256, 0, stream>>>(P2, G, coeff);       // P2 = newE (f32)

    // ---- new_E = LN(relu(newE @ W_Eout)*emask + E) --------------------
    k_cvt_f16  <<<ew(NED), 256, 0, stream>>>(P2, P2h, NED);
    k_cvt_f16_T<<<ew(DE * DE), 256, 0, stream>>>(W_Eout, WEt, DE, DE, DE * DE);
    {
        dim3 g(DE / 32, (B * NN) / 32, 1);                    // (2, 8192, 1)
        wmma_gemm_f16t<<<g, 32, 0, stream>>>(P2h, WEt, out_E, DE, DE, DE, DE,
                                             0, 0, 0);
    }
    k_newE_ln<<<B * NN, DE, 0, stream>>>(out_E, E, nm, ge, be);

    // ---- global readouts ----------------------------------------------
    k_g_enc<<<B * GK, 256, 0, stream>>>(E, nm, G_ker, Genc);
    k_x_enc<<<B * XK, N, 0, stream>>>(X, nm, X_ker, Xenc);
    gemm(stream, Genc, W_Gtr, g1, B, DY, GK * DE, GK * DE, DY, DY);
    gemm(stream, Xenc, W_Xtr, x1, B, DY, XK * DX, XK * DX, DY, DY);
    k_combine<<<ew(B * DY), 256, 0, stream>>>(g1, x1, hbuf);
    gemm(stream, hbuf, W_yout, t5, B, DY, DY, DY, DY, DY);
    k_newy_ln<<<B, DY, 0, stream>>>(t5, y, gy, by, out_y);
}
